// MySelfAttention_25366076850403
// MI455X (gfx1250) — matrix operand order repaired and checked by reading, not compiled, not run on hardware
//
#include <hip/hip_runtime.h>
#include <hip/hip_bf16.h>

typedef __attribute__((ext_vector_type(16))) __bf16 v16bf;
typedef __attribute__((ext_vector_type(8)))  float  v8f;

#define D_DIM   1024
#define B_DIM   4
#define T_DIM   4096
#define BT_DIM  16384
#define INV_SQRT_D 0.03125f

union Frag {
  uint4 u[2];
  v16bf v;
};

__device__ __forceinline__ unsigned short f2bf(float f) {
  unsigned u = __float_as_uint(f);
  unsigned r = (u + 0x7FFFu + ((u >> 16) & 1u)) >> 16;  // RNE
  return (unsigned short)r;
}
__device__ __forceinline__ unsigned pk2(float a, float b) {
  return (unsigned)f2bf(a) | ((unsigned)f2bf(b) << 16);
}
__device__ __forceinline__ v8f wmma_bf16(const Frag& a, const Frag& b, v8f c) {
  return __builtin_amdgcn_wmma_f32_16x16x32_bf16(false, a.v, false, b.v,
                                                 (short)0, c, false, false);
}

// ---------------- Kernel 1: QKV projection  C[m,n] = sum_k x[m,k]*W[n,k] + b[n]
__global__ __launch_bounds__(256, 1)
void qkv_kernel(const float* __restrict__ x,
                const float* __restrict__ Wq, const float* __restrict__ bq,
                const float* __restrict__ Wk, const float* __restrict__ bk,
                const float* __restrict__ Wv, const float* __restrict__ bv,
                unsigned short* __restrict__ Qb, unsigned short* __restrict__ Kb,
                unsigned short* __restrict__ Vt)
{
  constexpr int LDSTR = 40;  // bf16 elems per LDS row (32 data + 8 pad)
  __shared__ unsigned short sA[128 * LDSTR];
  __shared__ unsigned short sB[128 * LDSTR];

  const int z = blockIdx.z;
  const float* W    = (z == 0) ? Wq : (z == 1) ? Wk : Wv;
  const float* bias = (z == 0) ? bq : (z == 1) ? bk : bv;
  unsigned short* outp = (z == 0) ? Qb : (z == 1) ? Kb : Vt;
  const bool isV = (z == 2);

  const int tid = threadIdx.x;
  const int wid = tid >> 5, lane = tid & 31;
  const int hf = lane >> 4, mr = lane & 15;
  const int wm = wid & 3, wn = wid >> 2;     // 4x2 wave grid

  const int m0 = blockIdx.x * 128;
  const int n0 = blockIdx.y * 128;

  const int lrow = tid >> 1;            // 0..127
  const int lcol = (tid & 1) * 16;      // 0 or 16

  v8f acc[2][4];
#pragma unroll
  for (int i = 0; i < 2; ++i)
#pragma unroll
    for (int j = 0; j < 4; ++j) acc[i][j] = (v8f){0,0,0,0,0,0,0,0};

  for (int k0 = 0; k0 < D_DIM; k0 += 32) {
    const float4* ga = (const float4*)(x + (size_t)(m0 + lrow) * D_DIM + k0 + lcol);
    float4 a0 = ga[0], a1 = ga[1], a2 = ga[2], a3 = ga[3];
    const float4* gb = (const float4*)(W + (size_t)(n0 + lrow) * D_DIM + k0 + lcol);
    float4 b0 = gb[0], b1 = gb[1], b2 = gb[2], b3 = gb[3];

    __syncthreads();  // previous iteration's LDS reads complete
    uint4 pa0 = make_uint4(pk2(a0.x,a0.y), pk2(a0.z,a0.w), pk2(a1.x,a1.y), pk2(a1.z,a1.w));
    uint4 pa1 = make_uint4(pk2(a2.x,a2.y), pk2(a2.z,a2.w), pk2(a3.x,a3.y), pk2(a3.z,a3.w));
    *(uint4*)&sA[lrow * LDSTR + lcol]     = pa0;
    *(uint4*)&sA[lrow * LDSTR + lcol + 8] = pa1;
    uint4 pb0 = make_uint4(pk2(b0.x,b0.y), pk2(b0.z,b0.w), pk2(b1.x,b1.y), pk2(b1.z,b1.w));
    uint4 pb1 = make_uint4(pk2(b2.x,b2.y), pk2(b2.z,b2.w), pk2(b3.x,b3.y), pk2(b3.z,b3.w));
    *(uint4*)&sB[lrow * LDSTR + lcol]     = pb0;
    *(uint4*)&sB[lrow * LDSTR + lcol + 8] = pb1;
    __syncthreads();

    Frag af[2], bf[4];
#pragma unroll
    for (int mt = 0; mt < 2; ++mt) {
      const int row = wm * 32 + mt * 16 + mr;
      af[mt].u[0] = *(const uint4*)&sA[row * LDSTR + 16 * hf];       // K = 16h..16h+7
      af[mt].u[1] = *(const uint4*)&sA[row * LDSTR + 16 * hf + 8];   // K = 16h+8..16h+15
    }
#pragma unroll
    for (int nt = 0; nt < 4; ++nt) {
      const int row = wn * 64 + nt * 16 + mr;
      const uint4* p = (const uint4*)&sB[row * LDSTR + 16 * hf];     // K = 16h..16h+15
      bf[nt].u[0] = p[0];
      bf[nt].u[1] = p[1];
    }
#pragma unroll
    for (int mt = 0; mt < 2; ++mt)
#pragma unroll
      for (int nt = 0; nt < 4; ++nt)
        acc[mt][nt] = wmma_bf16(af[mt], bf[nt], acc[mt][nt]);
  }

  // Epilogue: uniform branch hoisted; V case packs 8 contiguous rows per lane.
  if (!isV) {
#pragma unroll
    for (int nt = 0; nt < 4; ++nt) {
      const int gn = n0 + wn * 64 + nt * 16 + mr;
      const float bn = bias[gn];
#pragma unroll
      for (int mt = 0; mt < 2; ++mt)
#pragma unroll
        for (int v = 0; v < 8; ++v) {
          const int gm = m0 + wm * 32 + mt * 16 + v + 8 * hf;
          outp[(size_t)gm * D_DIM + gn] = f2bf(acc[mt][nt][v] + bn);
        }
    }
  } else {
#pragma unroll
    for (int nt = 0; nt < 4; ++nt) {
      const int gn = n0 + wn * 64 + nt * 16 + mr;
      const float bn = bias[gn];
#pragma unroll
      for (int mt = 0; mt < 2; ++mt) {
        const int gmb = m0 + wm * 32 + mt * 16 + 8 * hf;   // 16B aligned
        const v8f a = acc[mt][nt];
        uint4 pv = make_uint4(pk2(a[0] + bn, a[1] + bn), pk2(a[2] + bn, a[3] + bn),
                              pk2(a[4] + bn, a[5] + bn), pk2(a[6] + bn, a[7] + bn));
        *(uint4*)&outp[(size_t)gn * BT_DIM + gmb] = pv;    // Vt[gn, gmb..gmb+7]
      }
    }
  }
}

// ---------------- Kernel 2: online softmax stats (row max m, sum-exp l)
__global__ __launch_bounds__(256, 1)
void attn_stats_kernel(const unsigned short* __restrict__ Qb,
                       const unsigned short* __restrict__ Kb,
                       float* __restrict__ mbuf, float* __restrict__ lbuf)
{
  const int tid = threadIdx.x;
  const int wid = tid >> 5, lane = tid & 31;
  const int hf = lane >> 4, mr = lane & 15;
  const int b = blockIdx.y;
  const int q0 = blockIdx.x * 128 + wid * 16;

  // Register-resident Q fragments for this wave's 16 rows (full depth 1024)
  const unsigned short* qbase = Qb + (size_t)(b * T_DIM + q0 + mr) * D_DIM;
  Frag qf[32];
#pragma unroll
  for (int s = 0; s < 32; ++s) {
    qf[s].u[0] = *(const uint4*)(qbase + 32 * s + 16 * hf);
    qf[s].u[1] = *(const uint4*)(qbase + 32 * s + 16 * hf + 8);
  }

  float mrun[8], lrun[8];
#pragma unroll
  for (int v = 0; v < 8; ++v) { mrun[v] = -1e30f; lrun[v] = 0.f; }

  for (int kt = 0; kt < T_DIM; kt += 16) {
    const unsigned short* kbase =
        Kb + (size_t)(b * T_DIM + kt + mr) * D_DIM + 16 * hf;
    if (kt + 16 < T_DIM)
      __builtin_prefetch(kbase + D_DIM * 16, 0, 1);   // next tile's K row
    v8f acc = (v8f){0,0,0,0,0,0,0,0};
#pragma unroll
    for (int s = 0; s < 32; ++s) {
      Frag kf;
      kf.u[0] = *(const uint4*)(kbase + 32 * s);
      kf.u[1] = *(const uint4*)(kbase + 32 * s + 8);
      acc = wmma_bf16(qf[s], kf, acc);
    }
#pragma unroll
    for (int v = 0; v < 8; ++v) {
      float sv = acc[v] * INV_SQRT_D;          // row = v+8h, col = mr
      float rm = sv;
      rm = fmaxf(rm, __shfl_xor(rm, 1, 32));
      rm = fmaxf(rm, __shfl_xor(rm, 2, 32));
      rm = fmaxf(rm, __shfl_xor(rm, 4, 32));
      rm = fmaxf(rm, __shfl_xor(rm, 8, 32));   // row max over 16 cols
      const float nm = fmaxf(mrun[v], rm);
      float e = __expf(sv - nm);
      e += __shfl_xor(e, 1, 32);
      e += __shfl_xor(e, 2, 32);
      e += __shfl_xor(e, 4, 32);
      e += __shfl_xor(e, 8, 32);               // row sum of exp
      lrun[v] = lrun[v] * __expf(mrun[v] - nm) + e;
      mrun[v] = nm;
    }
  }
  if (mr == 0) {
#pragma unroll
    for (int v = 0; v < 8; ++v) {
      const int row = q0 + v + 8 * hf;
      mbuf[b * T_DIM + row] = mrun[v];
      lbuf[b * T_DIM + row] = lrun[v];
    }
  }
}

// ---------------- Kernel 3: out = (exp(S - m)/l) @ V
__global__ __launch_bounds__(256, 1)
void attn_out_kernel(const unsigned short* __restrict__ Qb,
                     const unsigned short* __restrict__ Kb,
                     const unsigned short* __restrict__ Vt,
                     const float* __restrict__ mbuf, const float* __restrict__ lbuf,
                     float* __restrict__ out)
{
  __shared__ unsigned short sP[16 * 136];    // 16 q rows x 128 k cols, +8 pad

  const int tid = threadIdx.x;
  const int wid = tid >> 5, lane = tid & 31;
  const int hf = lane >> 4, mr = lane & 15;
  const int b = blockIdx.y;
  const int q0 = blockIdx.x * 16;

  const unsigned short* qbase = Qb + (size_t)(b * T_DIM + q0 + mr) * D_DIM;
  Frag qf[32];
#pragma unroll
  for (int s = 0; s < 32; ++s) {
    qf[s].u[0] = *(const uint4*)(qbase + 32 * s + 16 * hf);
    qf[s].u[1] = *(const uint4*)(qbase + 32 * s + 16 * hf + 8);
  }

  float mrow[8], linv[8];
#pragma unroll
  for (int v = 0; v < 8; ++v) {
    const int row = q0 + v + 8 * hf;
    mrow[v] = mbuf[b * T_DIM + row];
    linv[v] = 1.f / lbuf[b * T_DIM + row];
  }

  v8f o[8];
#pragma unroll
  for (int dt = 0; dt < 8; ++dt) o[dt] = (v8f){0,0,0,0,0,0,0,0};

  for (int kc = 0; kc < T_DIM; kc += 128) {
    // this wave recomputes scores for k columns [kc+wid*16, kc+wid*16+16)
    const unsigned short* kbase =
        Kb + (size_t)(b * T_DIM + kc + wid * 16 + mr) * D_DIM + 16 * hf;
    if (kc + 128 < T_DIM)
      __builtin_prefetch(kbase + D_DIM * 128, 0, 1);  // next chunk's K row
    v8f acc = (v8f){0,0,0,0,0,0,0,0};
#pragma unroll
    for (int s = 0; s < 32; ++s) {
      Frag kf;
      kf.u[0] = *(const uint4*)(kbase + 32 * s);
      kf.u[1] = *(const uint4*)(kbase + 32 * s + 8);
      acc = wmma_bf16(qf[s], kf, acc);
    }
    __syncthreads();  // previous iteration's sP reads complete
#pragma unroll
    for (int v = 0; v < 8; ++v) {
      const float p = __expf(acc[v] * INV_SQRT_D - mrow[v]) * linv[v];
      sP[(v + 8 * hf) * 136 + wid * 16 + mr] = f2bf(p);
    }
    __syncthreads();

#pragma unroll
    for (int kk = 0; kk < 4; ++kk) {
      Frag pf;
      const unsigned short* pr = sP + mr * 136 + kk * 32;
      pf.u[0] = *(const uint4*)(pr + 16 * hf);
      pf.u[1] = *(const uint4*)(pr + 16 * hf + 8);
#pragma unroll
      for (int dt = 0; dt < 8; ++dt) {
        const int dcol = wid * 128 + dt * 16 + mr;
        const unsigned short* vp =
            Vt + (size_t)dcol * BT_DIM + (b * T_DIM + kc + kk * 32 + 16 * hf);
        Frag vf;
        vf.u[0] = *(const uint4*)vp;
        vf.u[1] = *(const uint4*)(vp + 8);
        o[dt] = wmma_bf16(pf, vf, o[dt]);
      }
    }
  }

#pragma unroll
  for (int dt = 0; dt < 8; ++dt) {
    const int dcol = wid * 128 + dt * 16 + mr;
#pragma unroll
    for (int v = 0; v < 8; ++v) {
      const int row = q0 + v + 8 * hf;
      out[((size_t)b * T_DIM + row) * D_DIM + dcol] = o[dt][v];
    }
  }
}

extern "C" void kernel_launch(void* const* d_in, const int* in_sizes, int n_in,
                              void* d_out, int out_size, void* d_ws, size_t ws_size,
                              hipStream_t stream) {
  const float* x  = (const float*)d_in[0];
  const float* Wq = (const float*)d_in[1];
  const float* bq = (const float*)d_in[2];
  const float* Wk = (const float*)d_in[3];
  const float* bk = (const float*)d_in[4];
  const float* Wv = (const float*)d_in[5];
  const float* bv = (const float*)d_in[6];
  float* out = (float*)d_out;

  unsigned short* Qb = (unsigned short*)d_ws;
  unsigned short* Kb = Qb + (size_t)BT_DIM * D_DIM;
  unsigned short* Vt = Kb + (size_t)BT_DIM * D_DIM;
  float* mbuf = (float*)(Vt + (size_t)BT_DIM * D_DIM);
  float* lbuf = mbuf + BT_DIM;

  qkv_kernel<<<dim3(BT_DIM / 128, D_DIM / 128, 3), 256, 0, stream>>>(
      x, Wq, bq, Wk, bk, Wv, bv, Qb, Kb, Vt);
  attn_stats_kernel<<<dim3(T_DIM / 128, B_DIM), 256, 0, stream>>>(
      Qb, Kb, mbuf, lbuf);
  attn_out_kernel<<<dim3(T_DIM / 16, B_DIM), 256, 0, stream>>>(
      Qb, Kb, Vt, mbuf, lbuf, out);
}
